// NodeLayer_38216619000477
// MI455X (gfx1250) — matrix operand order repaired and checked by reading, not compiled, not run on hardware
//
#include <hip/hip_runtime.h>
#include <hip/hip_bf16.h>
#include <math.h>

typedef __attribute__((ext_vector_type(16))) __bf16 v16bf;
typedef __attribute__((ext_vector_type(8)))  __bf16 v8bf;
typedef __attribute__((ext_vector_type(8)))  float  v8f;

static __device__ __forceinline__ __bf16 f32_to_bf16_rne(float x) {
    unsigned u = __builtin_bit_cast(unsigned, x);
    u += 0x7FFFu + ((u >> 16) & 1u);
    unsigned short h = (unsigned short)(u >> 16);
    return __builtin_bit_cast(__bf16, h);
}
static __device__ __forceinline__ float bf16_to_f32(__bf16 b) {
    unsigned u = ((unsigned)__builtin_bit_cast(unsigned short, b)) << 16;
    return __builtin_bit_cast(float, u);
}
static __device__ __forceinline__ void split2(float x, __bf16& hi, __bf16& lo) {
    hi = f32_to_bf16_rne(x);
    lo = f32_to_bf16_rne(x - bf16_to_f32(hi));
}
static __device__ __forceinline__ v8f wmma_bf16(v16bf a, v16bf b, v8f c) {
    return __builtin_amdgcn_wmma_f32_16x16x32_bf16(false, a, false, b, (short)0, c,
                                                   false, false);
}
// Async global->LDS copy of 2x16B (hi/lo splits), tracked by ASYNCcnt.
static __device__ __forceinline__ void stage_async(const __bf16* gh, const __bf16* gl,
                                                   __bf16* lh, __bf16* ll) {
    asm volatile(
        "global_load_async_to_lds_b128 %0, %2, off\n\t"
        "global_load_async_to_lds_b128 %1, %3, off"
        :
        : "v"((unsigned)(size_t)lh), "v"((unsigned)(size_t)ll), "v"(gh), "v"(gl)
        : "memory");
}
// Lightweight workgroup barrier: no hardware counter drain (unlike
// __syncthreads, which forces s_wait_loadcnt_dscnt 0x0). Ordering of our LDS
// ring is already guaranteed by s_wait_asynccnt + WMMA data dependencies.
static __device__ __forceinline__ void wg_barrier() {
    asm volatile("s_barrier_signal -1\n\ts_barrier_wait -1" ::: "memory");
}

// Elementwise fp32 -> split bf16 (used once for x at t=0).
__global__ __launch_bounds__(256) void split_f32(
    const float* __restrict__ X, __bf16* __restrict__ Xhi,
    __bf16* __restrict__ Xlo, int total) {
    int i = blockIdx.x * 256 + threadIdx.x;
    if (i >= total) return;
    __bf16 h, l;
    split2(X[i], h, l);
    Xhi[i] = h;
    Xlo[i] = l;
}

// W[K][N] fp32 -> split-bf16 in WMMA fragment-linear order.
// piece addr (elements) = (((t*CPK + c)*2 + p)*32 + lane)*8
// element (lane, e=p*8+j): n = t*16 + lane%16, k = c*32 + (lane/16)*16 + e
__global__ __launch_bounds__(256) void swizzle_weights(
    const float* __restrict__ W, __bf16* __restrict__ Fhi,
    __bf16* __restrict__ Flo, int K, int N) {
    const int CPK = K >> 5;
    int f = blockIdx.x * 256 + threadIdx.x;  // piece index
    if (f >= (K * N) / 8) return;
    const int lane = f & 31;
    const int p = (f >> 5) & 1;
    const int g = f >> 6;
    const int c = g % CPK;
    const int t = g / CPK;
    const int n = t * 16 + (lane & 15);
    const int kbase = c * 32 + ((lane >> 4) << 4) + p * 8;
    const size_t o = (size_t)f * 8;
#pragma unroll
    for (int j = 0; j < 8; ++j) {
        float x = W[(size_t)(kbase + j) * N + n];
        __bf16 h, l;
        split2(x, h, l);
        Fhi[o + j] = h;
        Flo[o + j] = l;
    }
}

// C = A*B + bias with fused RK4 epilogue; bf16x3 split precision.
// A: split-bf16 [M][KD] row-major, staged to LDS via async-to-LDS 3-slot ring.
// B: frag-linear split-bf16, loaded straight to fragments.
template <int KD>
__global__ __launch_bounds__(256) void gemm_rk4(
    const __bf16* __restrict__ Ahi, const __bf16* __restrict__ Alo,
    const __bf16* __restrict__ BhiF, const __bf16* __restrict__ BloF,
    const float* __restrict__ bias,
    int N, int mode, float step_c, float acc_w, float dt6,
    __bf16* __restrict__ Ohi, __bf16* __restrict__ Olo,
    const float* __restrict__ z, float* __restrict__ accb,
    float* __restrict__ zf32) {

    constexpr int CPK = KD >> 5;
    __shared__ __attribute__((aligned(16))) __bf16 Ah[3][64][32];
    __shared__ __attribute__((aligned(16))) __bf16 Al[3][64][32];

    const int tid  = threadIdx.x;
    const int lane = tid & 31;
    const int wave = tid >> 5;
    const int wm   = wave >> 1;                 // 4 M-tiles of 16 -> BM=64
    const int wnn  = wave & 1;                  // 2 N-groups x 2 tiles -> BN=64
    const int m0   = blockIdx.y * 64;
    const int t0   = blockIdx.x * 4 + wnn * 2;  // first of 2 N-tiles

    const int srow = tid >> 2, sk = (tid & 3) * 8;  // staging: 16B/split/thread
    const __bf16* gA = Ahi + (size_t)(m0 + srow) * KD + sk;
    const __bf16* gL = Alo + (size_t)(m0 + srow) * KD + sk;

    const int ar  = wm * 16 + (lane & 15);
    const int akb = (lane < 16) ? 0 : 16;
    const __bf16* pB0h = BhiF + ((size_t)t0 * CPK) * 512 + (size_t)lane * 8;
    const __bf16* pB0l = BloF + ((size_t)t0 * CPK) * 512 + (size_t)lane * 8;
    const __bf16* pB1h = pB0h + (size_t)CPK * 512;
    const __bf16* pB1l = pB0l + (size_t)CPK * 512;

    v8f acc0 = {}, acc1 = {};

    auto body = [&](int c, int buf, bool do_stage, int nb) {
        // A fragment (contiguous k = 16*(lane/16) + e, matching B frag-linear)
        v8bf a0h = *(const v8bf*)&Ah[buf][ar][akb];
        v8bf a1h = *(const v8bf*)&Ah[buf][ar][akb + 8];
        v8bf a0l = *(const v8bf*)&Al[buf][ar][akb];
        v8bf a1l = *(const v8bf*)&Al[buf][ar][akb + 8];
        v16bf afh = __builtin_shufflevector(a0h, a1h, 0,1,2,3,4,5,6,7,8,9,10,11,12,13,14,15);
        v16bf afl = __builtin_shufflevector(a0l, a1l, 0,1,2,3,4,5,6,7,8,9,10,11,12,13,14,15);

        if (do_stage)
            stage_async(gA + (size_t)(c + 2) * 32, gL + (size_t)(c + 2) * 32,
                        &Ah[nb][srow][sk], &Al[nb][srow][sk]);

        // B fragments straight from global (frag-linear, coalesced b128)
        const size_t co = (size_t)c * 512;
        v8bf q0h = *(const v8bf*)(pB0h + co);
        v8bf q1h = *(const v8bf*)(pB0h + co + 256);
        v8bf q0l = *(const v8bf*)(pB0l + co);
        v8bf q1l = *(const v8bf*)(pB0l + co + 256);
        v8bf r0h = *(const v8bf*)(pB1h + co);
        v8bf r1h = *(const v8bf*)(pB1h + co + 256);
        v8bf r0l = *(const v8bf*)(pB1l + co);
        v8bf r1l = *(const v8bf*)(pB1l + co + 256);
        v16bf bf0h = __builtin_shufflevector(q0h, q1h, 0,1,2,3,4,5,6,7,8,9,10,11,12,13,14,15);
        v16bf bf0l = __builtin_shufflevector(q0l, q1l, 0,1,2,3,4,5,6,7,8,9,10,11,12,13,14,15);
        v16bf bf1h = __builtin_shufflevector(r0h, r1h, 0,1,2,3,4,5,6,7,8,9,10,11,12,13,14,15);
        v16bf bf1l = __builtin_shufflevector(r0l, r1l, 0,1,2,3,4,5,6,7,8,9,10,11,12,13,14,15);

        if (do_stage) {
            __builtin_prefetch(pB0h + co + 512, 0, 3);  // WGP-scope prefetch
            __builtin_prefetch(pB0l + co + 512, 0, 3);
            __builtin_prefetch(pB1h + co + 512, 0, 3);
            __builtin_prefetch(pB1l + co + 512, 0, 3);
        }

        // bf16x3 split precision, two N-tiles interleaved
        acc0 = wmma_bf16(afh, bf0h, acc0);
        acc1 = wmma_bf16(afh, bf1h, acc1);
        acc0 = wmma_bf16(afh, bf0l, acc0);
        acc1 = wmma_bf16(afh, bf1l, acc1);
        acc0 = wmma_bf16(afl, bf0h, acc0);
        acc1 = wmma_bf16(afl, bf1h, acc1);
    };

    // prologue: stage chunks 0 and 1
    stage_async(gA, gL, &Ah[0][srow][sk], &Al[0][srow][sk]);
    stage_async(gA + 32, gL + 32, &Ah[1][srow][sk], &Al[1][srow][sk]);

    // main loop: branch-free body, unconditional waits, carried ring slots
    int bufc = 0, bufs = 2;
    for (int c = 0; c < CPK - 2; ++c) {
        asm volatile("s_wait_asynccnt 0x2" ::: "memory");  // chunk c landed
        wg_barrier();
        body(c, bufc, true, bufs);
        bufc = (bufc == 2) ? 0 : bufc + 1;
        bufs = (bufs == 2) ? 0 : bufs + 1;
    }
    // tail: c = CPK-2 (chunk CPK-1 still in flight), then c = CPK-1
    asm volatile("s_wait_asynccnt 0x2" ::: "memory");
    wg_barrier();
    body(CPK - 2, bufc, false, 0);
    bufc = (bufc == 2) ? 0 : bufc + 1;
    asm volatile("s_wait_asynccnt 0x0" ::: "memory");
    wg_barrier();
    body(CPK - 1, bufc, false, 0);

    // epilogue: C layout -> VGPR r: M = r + (lane<16?0:8), N = lane%16
    const int nl = lane & 15;
    const int mh = (lane >> 4) * 8;
#pragma unroll
    for (int jt = 0; jt < 2; ++jt) {
        const int ng = (t0 + jt) * 16 + nl;
        const float bns = bias[ng];
#pragma unroll
        for (int r = 0; r < 8; ++r) {
            const int mg = m0 + wm * 16 + mh + r;
            const size_t idx = (size_t)mg * N + ng;
            const float v = (jt ? acc1[r] : acc0[r]) + bns;
            __bf16 h, l;
            if (mode == 0) {
                split2(tanhf(v), h, l);
                Ohi[idx] = h; Olo[idx] = l;
            } else if (mode == 1) {
                accb[idx] = v;
                split2(z[idx] + step_c * v, h, l);
                Ohi[idx] = h; Olo[idx] = l;
            } else if (mode == 4) {
                float zn = z[idx] + dt6 * (accb[idx] + v);
                zf32[idx] = zn;
                split2(zn, h, l);
                Ohi[idx] = h; Olo[idx] = l;
            } else {
                accb[idx] += acc_w * v;
                split2(z[idx] + step_c * v, h, l);
                Ohi[idx] = h; Olo[idx] = l;
            }
        }
    }
}

extern "C" void kernel_launch(void* const* d_in, const int* in_sizes, int n_in,
                              void* d_out, int out_size, void* d_ws, size_t ws_size,
                              hipStream_t stream) {
    const int B = 512, D = 1024, H = 2048;
    const float DT = 0.1f;

    const float* x  = (const float*)d_in[0];
    const float* W1 = (const float*)d_in[1];
    const float* b1 = (const float*)d_in[2];
    const float* W2 = (const float*)d_in[3];
    const float* b2 = (const float*)d_in[4];
    float* out = (float*)d_out;

    char* ws = (char*)d_ws;
    size_t off = 0;
    auto alloc = [&](size_t bytes) -> char* {
        char* p = ws + off;
        off += (bytes + 255) & ~(size_t)255;
        return p;
    };
    __bf16* W1hiF = (__bf16*)alloc((size_t)D * H * 2);
    __bf16* W1loF = (__bf16*)alloc((size_t)D * H * 2);
    __bf16* W2hiF = (__bf16*)alloc((size_t)D * H * 2);
    __bf16* W2loF = (__bf16*)alloc((size_t)D * H * 2);
    __bf16* Hhi   = (__bf16*)alloc((size_t)B * H * 2);
    __bf16* Hlo   = (__bf16*)alloc((size_t)B * H * 2);
    __bf16* Zhi   = (__bf16*)alloc((size_t)B * D * 2);
    __bf16* Zlo   = (__bf16*)alloc((size_t)B * D * 2);
    float*  zbuf  = (float*)alloc((size_t)B * D * 4);
    float*  accb  = (float*)alloc((size_t)B * D * 4);

    split_f32<<<(B * D + 255) / 256, 256, 0, stream>>>(x, Zhi, Zlo, B * D);
    const int wPieces = (D * H) / 8;
    swizzle_weights<<<(wPieces + 255) / 256, 256, 0, stream>>>(W1, W1hiF, W1loF, D, H);
    swizzle_weights<<<(wPieces + 255) / 256, 256, 0, stream>>>(W2, W2hiF, W2loF, H, D);

    dim3 blk(256);
    dim3 gridA(H / 64, B / 64);   // hidden layer: N=2048 -> 32x8 blocks
    dim3 gridB(D / 64, B / 64);   // output layer: N=1024 -> 16x8 blocks
    const float dt6 = DT / 6.0f;

    for (int s = 0; s < 10; ++s) {
        const float* zs = (s == 0) ? x : zbuf;
        float* znext    = (s == 9) ? out : zbuf;
        // k1
        gemm_rk4<1024><<<gridA, blk, 0, stream>>>(Zhi, Zlo, W1hiF, W1loF, b1, H, 0,
                                                  0.f, 0.f, 0.f, Hhi, Hlo, nullptr, nullptr, nullptr);
        gemm_rk4<2048><<<gridB, blk, 0, stream>>>(Hhi, Hlo, W2hiF, W2loF, b2, D, 1,
                                                  0.5f * DT, 1.0f, 0.f, Zhi, Zlo, zs, accb, nullptr);
        // k2
        gemm_rk4<1024><<<gridA, blk, 0, stream>>>(Zhi, Zlo, W1hiF, W1loF, b1, H, 0,
                                                  0.f, 0.f, 0.f, Hhi, Hlo, nullptr, nullptr, nullptr);
        gemm_rk4<2048><<<gridB, blk, 0, stream>>>(Hhi, Hlo, W2hiF, W2loF, b2, D, 2,
                                                  0.5f * DT, 2.0f, 0.f, Zhi, Zlo, zs, accb, nullptr);
        // k3
        gemm_rk4<1024><<<gridA, blk, 0, stream>>>(Zhi, Zlo, W1hiF, W1loF, b1, H, 0,
                                                  0.f, 0.f, 0.f, Hhi, Hlo, nullptr, nullptr, nullptr);
        gemm_rk4<2048><<<gridB, blk, 0, stream>>>(Hhi, Hlo, W2hiF, W2loF, b2, D, 3,
                                                  DT, 2.0f, 0.f, Zhi, Zlo, zs, accb, nullptr);
        // k4 + state update
        gemm_rk4<1024><<<gridA, blk, 0, stream>>>(Zhi, Zlo, W1hiF, W1loF, b1, H, 0,
                                                  0.f, 0.f, 0.f, Hhi, Hlo, nullptr, nullptr, nullptr);
        gemm_rk4<2048><<<gridB, blk, 0, stream>>>(Hhi, Hlo, W2hiF, W2loF, b2, D, 4,
                                                  0.f, 0.f, dt6, Zhi, Zlo, zs, accb, znext);
    }
}
